// Swin_TNT_block_37632503447915
// MI455X (gfx1250) — matrix operand order repaired and checked by reading, not compiled, not run on hardware
//
#include <hip/hip_runtime.h>
#include <hip/hip_bf16.h>
#include <math.h>
#include <stdint.h>

// ---------------------------------------------------------------- types
typedef __attribute__((ext_vector_type(16))) __bf16 v16bf;
typedef __attribute__((ext_vector_type(8)))  __bf16 v8bf;
typedef __attribute__((ext_vector_type(8)))  float  v8f;
typedef __attribute__((ext_vector_type(4)))  int    v4i;

union Frag { v16bf v; v8bf h[2]; };

__device__ __forceinline__ v8f wmma_bf16(const Frag& a, const Frag& b, v8f c) {
  return __builtin_amdgcn_wmma_f32_16x16x32_bf16(
      /*neg_a=*/false, a.v, /*neg_b=*/false, b.v,
      /*c_mod=*/(short)0, c, /*reuse_a=*/false, /*reuse_b=*/false);
}

// ---------------------------------------------------------------- async copy to LDS (CDNA5)
#if defined(__gfx1250__) && __has_builtin(__builtin_amdgcn_global_load_async_to_lds_b128)
#define ASYNC_LDS 1
typedef __attribute__((address_space(1))) v4i GV4;   // global v4i
typedef __attribute__((address_space(3))) v4i LV4;   // LDS v4i
#else
#define ASYNC_LDS 0
#endif

__device__ __forceinline__ void stage16(const __bf16* __restrict src, __bf16* dst) {
#if ASYNC_LDS
  __builtin_amdgcn_global_load_async_to_lds_b128(
      (GV4*)(uintptr_t)src, (LV4*)(uint32_t)(uintptr_t)dst, 0, 0);
#else
  *(v8bf*)dst = *(const v8bf*)src;
#endif
}

__device__ __forceinline__ void async_wait0() {
#if defined(__gfx1250__)
#if __has_builtin(__builtin_amdgcn_s_wait_asynccnt)
  __builtin_amdgcn_s_wait_asynccnt(0);
#else
  asm volatile("s_wait_asynccnt 0" ::: "memory");
#endif
#endif
}

// ---------------------------------------------------------------- sizes
// B=16 H=W=112 C=384 NH=12 HD=32 WS=7 N=49 (pad->64) NW=256 BNW=4096
#define BIMG   16
#define CDIM   384
#define NHEAD  12
#define HD     32
#define NPAD   64
#define NWIN   256
#define BNW    4096            // BIMG*NWIN
#define NTOK   200704          // BIMG*112*112
#define MLPH   1536

// workspace layout (bytes)
static constexpr size_t OFF_QKVW = 0;                      // 1152x384 bf16
static constexpr size_t OFF_PROJW = 884736;                // 384x384 bf16
static constexpr size_t OFF_FC1W  = 1179648;               // 1536x384 bf16
static constexpr size_t OFF_FC2W  = 2359296;               // 384x1536 bf16
static constexpr size_t OFF_XW    = (size_t)4 << 20;       // BNW*64*384 bf16
static constexpr size_t SZ_XW     = (size_t)BNW * NPAD * CDIM * 2;
static constexpr size_t OFF_Q     = OFF_XW + SZ_XW;        // BNW*12*64*32 bf16
static constexpr size_t SZ_HEADS  = (size_t)BNW * NHEAD * NPAD * HD * 2;
static constexpr size_t OFF_K     = OFF_Q + SZ_HEADS;
static constexpr size_t OFF_VT    = OFF_K + SZ_HEADS;      // transposed: HD x N
static constexpr size_t OFF_O     = OFF_VT + SZ_HEADS;     // BNW*64*384 bf16
static constexpr size_t OFF_H2    = OFF_XW;                // alias (Xw dead)
static constexpr size_t OFF_G     = OFF_Q;                 // alias (QKV dead)

// ---------------------------------------------------------------- k0: W -> Wt (bf16, N-major)
__global__ void k0_convert(const float* __restrict W, __bf16* __restrict Wt,
                           int K, int N) {
  size_t total = (size_t)K * N;
  for (size_t i = (size_t)blockIdx.x * blockDim.x + threadIdx.x; i < total;
       i += (size_t)gridDim.x * blockDim.x) {
    size_t n = i / K, k = i % K;
    Wt[i] = (__bf16)W[k * (size_t)N + n];
  }
}

// ---------------------------------------------------------------- k1: LN1 + shift + window partition
__global__ void k1_ln_shift(const float* __restrict x,
                            const float* __restrict g, const float* __restrict b,
                            __bf16* __restrict Xw) {
  int blk = blockIdx.x;                 // b*256 + wy*16 + wx
  int bi = blk >> 8, wid = blk & 255;
  int wy = wid >> 4, wx = wid & 15;
  int wave = threadIdx.x >> 5, lane = threadIdx.x & 31;
  for (int n = wave; n < 49; n += 8) {
    int iy = n / 7, ix = n % 7;
    int yy = (wy * 7 + iy + 3) % 112;   // roll(-3)
    int xx = (wx * 7 + ix + 3) % 112;
    const float* row = x + ((size_t)bi * 12544 + yy * 112 + xx) * CDIM;
    float v[12]; float s = 0.f;
    #pragma unroll
    for (int i = 0; i < 12; ++i) { v[i] = row[lane + 32 * i]; s += v[i]; }
    #pragma unroll
    for (int o = 16; o > 0; o >>= 1) s += __shfl_xor(s, o);
    float mean = s * (1.f / 384.f);
    float var = 0.f;
    #pragma unroll
    for (int i = 0; i < 12; ++i) { float d = v[i] - mean; var += d * d; }
    #pragma unroll
    for (int o = 16; o > 0; o >>= 1) var += __shfl_xor(var, o);
    float inv = rsqrtf(var * (1.f / 384.f) + 1e-5f);
    __bf16* orow = Xw + ((size_t)blk * NPAD + n) * CDIM;
    #pragma unroll
    for (int i = 0; i < 12; ++i) {
      int c = lane + 32 * i;
      orow[c] = (__bf16)((v[i] - mean) * inv * g[c] + b[c]);
    }
  }
  // zero the 15 padded rows
  for (int t = threadIdx.x; t < 15 * CDIM; t += 256)
    Xw[((size_t)blk * NPAD + 49) * CDIM + t] = (__bf16)0.f;
}

// ---------------------------------------------------------------- generic bf16 WMMA GEMM tile
// block = 256 threads (8 waves). block tile: 128 rows x 64 cols.
// wave w: rows [128*bx + 16w, +16), cols [64*by, +64) -> 4 accumulators.
// B tile (64 cols x 32 K, 4KB) staged in LDS, double-buffered via async copy.
template <int KDIM, class Epi>
__device__ __forceinline__ void gemm_block(const __bf16* __restrict A,
                                           const __bf16* __restrict Wt, Epi epi) {
  __shared__ __align__(16) __bf16 Bs[2][64 * 32];
  const int lane = threadIdx.x & 31;
  const int wave = threadIdx.x >> 5;
  const int rowBase = blockIdx.x * 128 + wave * 16;
  const int colBase = blockIdx.y * 64;
  const int m = rowBase + (lane & 15);
  const int koff = (lane & 16) ? 16 : 0;         // A-frag K sub-offset per lane group
  const int g16 = (lane >> 4) * 16;              // B-frag K block per lane group
  v8f acc[4] = {};
  const __bf16* ap = A + (size_t)m * KDIM + koff;

  // staging: thread t copies 16B -> Bs[buf][col=t/4][ (t%4)*8 .. +7 ]
  const int sr = threadIdx.x >> 2;               // column within 64-wide tile
  const int sc = (threadIdx.x & 3) * 8;          // halfs within 32-deep K chunk
  const __bf16* srow = Wt + (size_t)(colBase + sr) * KDIM + sc;
  __bf16* sdst0 = &Bs[0][sr * 32 + sc];
  __bf16* sdst1 = &Bs[1][sr * 32 + sc];

  stage16(srow, sdst0);
  async_wait0();
  __syncthreads();

  int cur = 0;
  for (int kb = 0; kb < KDIM; kb += 32) {
    if (kb + 32 < KDIM)                           // prefetch next K-chunk into other buffer
      stage16(srow + kb + 32, cur ? sdst0 : sdst1);

    __builtin_prefetch(ap + kb + 128, 0, 2);
    Frag a;
    a.h[0] = *(const v8bf*)(ap + kb);             // K = koff..koff+7
    a.h[1] = *(const v8bf*)(ap + kb + 8);         // K = koff+8..koff+15
    const __bf16* bb = &Bs[cur][(lane & 15) * 32 + g16];
    #pragma unroll
    for (int t = 0; t < 4; ++t) {
      const __bf16* bp = bb + t * 16 * 32;
      Frag bf;
      bf.h[0] = *(const v8bf*)bp;                 // K = g16..g16+7
      bf.h[1] = *(const v8bf*)(bp + 8);           // K = g16+8..g16+15
      acc[t] = wmma_bf16(a, bf, acc[t]);
    }
    async_wait0();
    __syncthreads();
    cur ^= 1;
  }

  const int rofs = (lane >> 4) << 3;
  #pragma unroll
  for (int t = 0; t < 4; ++t)
    #pragma unroll
    for (int j = 0; j < 8; ++j)
      epi(rowBase + j + rofs, colBase + t * 16 + (lane & 15), acc[t][j]);
}

// ---------------------------------------------------------------- k2: QKV GEMM + head scatter
__global__ void k2_qkv(const __bf16* __restrict Xw, const __bf16* __restrict Wt,
                       const float* __restrict bias, __bf16* __restrict Q,
                       __bf16* __restrict Kh, __bf16* __restrict Vt) {
  gemm_block<CDIM>(Xw, Wt, [=](int row, int col, float v) {
    v += bias[col];
    int which = col / CDIM, c = col % CDIM, h = c >> 5, d = c & 31;
    int bnw = row >> 6, n = row & 63;
    size_t hr = (size_t)bnw * NHEAD + h;
    if (which == 0)      Q[(hr * NPAD + n) * HD + d] = (__bf16)(v * 0.17677669529663687f);
    else if (which == 1) Kh[(hr * NPAD + n) * HD + d] = (__bf16)v;
    else                 Vt[(hr * HD + d) * NPAD + n] = (__bf16)v;  // transposed
  });
}

// ---------------------------------------------------------------- k3: attention per (window, head)
__device__ __forceinline__ int sw_region(int p) { return p < 105 ? 0 : (p < 109 ? 1 : 2); }

__global__ void __launch_bounds__(128)
k3_attn(const __bf16* __restrict Q, const __bf16* __restrict Kh,
        const __bf16* __restrict Vt, const float* __restrict rpb,
        __bf16* __restrict O) {
  __shared__ float S[64][65];
  __shared__ __align__(16) __bf16 P[64][64];
  int blk = blockIdx.x;
  int h = blk % NHEAD, bnw = blk / NHEAD;
  int lane = threadIdx.x & 31, wave = threadIdx.x >> 5;    // 4 waves
  int koff = (lane & 16) ? 16 : 0;
  int g16 = (lane >> 4) * 16;
  const __bf16* Qw = Q  + ((size_t)bnw * NHEAD + h) * NPAD * HD;
  const __bf16* Kw = Kh + ((size_t)bnw * NHEAD + h) * NPAD * HD;
  const __bf16* Vw = Vt + ((size_t)bnw * NHEAD + h) * HD * NPAD;

  // ---- S = Q @ K^T  (4 waves x 4 n-tiles, K=HD=32 -> one WMMA per tile)
  {
    Frag a;
    const __bf16* ap = Qw + (size_t)(wave * 16 + (lane & 15)) * HD + koff;
    a.h[0] = *(const v8bf*)ap;
    a.h[1] = *(const v8bf*)(ap + 8);
    #pragma unroll
    for (int nt = 0; nt < 4; ++nt) {
      const __bf16* bp = Kw + (size_t)(nt * 16 + (lane & 15)) * HD + g16;
      Frag bf;
      bf.h[0] = *(const v8bf*)bp;
      bf.h[1] = *(const v8bf*)(bp + 8);
      v8f acc = {};
      acc = wmma_bf16(a, bf, acc);
      int rofs = (lane >> 4) << 3;
      #pragma unroll
      for (int j = 0; j < 8; ++j)
        S[wave * 16 + j + rofs][nt * 16 + (lane & 15)] = acc[j];
    }
  }
  __syncthreads();

  // ---- softmax rows (bias + shift-mask computed inline)
  if (threadIdx.x < 64) {
    int r = threadIdx.x;
    if (r < 49) {
      int wid = bnw & 255, wy = wid >> 4, wx = wid & 15;
      int iy = r / 7, ix = r % 7;
      int cr = 3 * sw_region(wy * 7 + iy) + sw_region(wx * 7 + ix);
      float vals[49]; float mx = -1e30f;
      for (int c = 0; c < 49; ++c) {
        int jy = c / 7, jx = c % 7;
        float bias = rpb[((iy - jy + 6) * 13 + (ix - jx + 6)) * NHEAD + h];
        int cc = 3 * sw_region(wy * 7 + jy) + sw_region(wx * 7 + jx);
        float v = S[r][c] + bias + ((cc != cr) ? -100.f : 0.f);
        vals[c] = v; mx = fmaxf(mx, v);
      }
      float sum = 0.f;
      for (int c = 0; c < 49; ++c) { float e = __expf(vals[c] - mx); vals[c] = e; sum += e; }
      float inv = 1.f / sum;
      for (int c = 0; c < 49; ++c) P[r][c] = (__bf16)(vals[c] * inv);
      for (int c = 49; c < 64; ++c) P[r][c] = (__bf16)0.f;   // pad cols kill V pad rows
    } else {
      for (int c = 0; c < 64; ++c) P[r][c] = (__bf16)0.f;
    }
  }
  __syncthreads();

  // ---- O = P @ V  (K=64 in 2 steps; B from transposed V = contiguous)
  #pragma unroll
  for (int nt = 0; nt < 2; ++nt) {
    v8f acc = {};
    #pragma unroll
    for (int kb = 0; kb < 64; kb += 32) {
      Frag a;
      const __bf16* ap = &P[wave * 16 + (lane & 15)][kb + koff];
      a.h[0] = *(const v8bf*)ap;
      a.h[1] = *(const v8bf*)(ap + 8);
      const __bf16* bp = Vw + (size_t)(nt * 16 + (lane & 15)) * NPAD + kb + g16;
      Frag bf;
      bf.h[0] = *(const v8bf*)bp;
      bf.h[1] = *(const v8bf*)(bp + 8);
      acc = wmma_bf16(a, bf, acc);
    }
    int rofs = (lane >> 4) << 3;
    #pragma unroll
    for (int j = 0; j < 8; ++j) {
      int m = wave * 16 + j + rofs;
      O[((size_t)bnw * NPAD + m) * CDIM + h * HD + nt * 16 + (lane & 15)] = (__bf16)acc[j];
    }
  }
}

// ---------------------------------------------------------------- k4: proj GEMM + window reverse + residual
__global__ void k4_proj(const __bf16* __restrict O, const __bf16* __restrict Wt,
                        const float* __restrict bias, const float* __restrict xin,
                        float* __restrict out) {
  gemm_block<CDIM>(O, Wt, [=](int row, int col, float v) {
    int bnw = row >> 6, n = row & 63;
    if (n >= 49) return;
    int bi = bnw >> 8, wid = bnw & 255;
    int wy = wid >> 4, wx = wid & 15;
    int iy = n / 7, ix = n % 7;
    int yy = (wy * 7 + iy + 3) % 112;   // roll(+3) on reverse
    int xx = (wx * 7 + ix + 3) % 112;
    size_t tok = (size_t)bi * 12544 + yy * 112 + xx;
    out[tok * CDIM + col] = xin[tok * CDIM + col] + v + bias[col];
  });
}

// ---------------------------------------------------------------- k5: LN2 -> bf16
__global__ void k5_ln2(const float* __restrict X1, const float* __restrict g,
                       const float* __restrict b, __bf16* __restrict H2) {
  int wave = threadIdx.x >> 5, lane = threadIdx.x & 31;
  size_t tok = (size_t)blockIdx.x * 8 + wave;
  const float* row = X1 + tok * CDIM;
  float v[12]; float s = 0.f;
  #pragma unroll
  for (int i = 0; i < 12; ++i) { v[i] = row[lane + 32 * i]; s += v[i]; }
  #pragma unroll
  for (int o = 16; o > 0; o >>= 1) s += __shfl_xor(s, o);
  float mean = s * (1.f / 384.f);
  float var = 0.f;
  #pragma unroll
  for (int i = 0; i < 12; ++i) { float d = v[i] - mean; var += d * d; }
  #pragma unroll
  for (int o = 16; o > 0; o >>= 1) var += __shfl_xor(var, o);
  float inv = rsqrtf(var * (1.f / 384.f) + 1e-5f);
  __bf16* orow = H2 + tok * CDIM;
  #pragma unroll
  for (int i = 0; i < 12; ++i) {
    int c = lane + 32 * i;
    orow[c] = (__bf16)((v[i] - mean) * inv * g[c] + b[c]);
  }
}

// ---------------------------------------------------------------- k6: FC1 GEMM + exact GELU
__global__ void k6_fc1(const __bf16* __restrict H2, const __bf16* __restrict Wt,
                       const float* __restrict bias, __bf16* __restrict G) {
  gemm_block<CDIM>(H2, Wt, [=](int row, int col, float v) {
    v += bias[col];
    float gl = 0.5f * v * (1.0f + erff(v * 0.70710678118654752f));
    G[(size_t)row * MLPH + col] = (__bf16)gl;
  });
}

// ---------------------------------------------------------------- k7: FC2 GEMM + residual into d_out
__global__ void k7_fc2(const __bf16* __restrict G, const __bf16* __restrict Wt,
                       const float* __restrict bias, float* __restrict out) {
  gemm_block<MLPH>(G, Wt, [=](int row, int col, float v) {
    out[(size_t)row * CDIM + col] += v + bias[col];
  });
}

// ---------------------------------------------------------------- host
extern "C" void kernel_launch(void* const* d_in, const int* in_sizes, int n_in,
                              void* d_out, int out_size, void* d_ws, size_t ws_size,
                              hipStream_t stream) {
  const float* x       = (const float*)d_in[0];
  const float* norm1_g = (const float*)d_in[1];
  const float* norm1_b = (const float*)d_in[2];
  const float* qkv_w   = (const float*)d_in[3];
  const float* qkv_b   = (const float*)d_in[4];
  const float* proj_w  = (const float*)d_in[5];
  const float* proj_b  = (const float*)d_in[6];
  const float* rpb     = (const float*)d_in[7];
  const float* norm2_g = (const float*)d_in[8];
  const float* norm2_b = (const float*)d_in[9];
  const float* fc1_w   = (const float*)d_in[10];
  const float* fc1_b   = (const float*)d_in[11];
  const float* fc2_w   = (const float*)d_in[12];
  const float* fc2_b   = (const float*)d_in[13];
  float* out = (float*)d_out;

  char* ws = (char*)d_ws;
  __bf16* qkvWt = (__bf16*)(ws + OFF_QKVW);
  __bf16* projWt = (__bf16*)(ws + OFF_PROJW);
  __bf16* fc1Wt = (__bf16*)(ws + OFF_FC1W);
  __bf16* fc2Wt = (__bf16*)(ws + OFF_FC2W);
  __bf16* Xw = (__bf16*)(ws + OFF_XW);
  __bf16* Qb = (__bf16*)(ws + OFF_Q);
  __bf16* Kb = (__bf16*)(ws + OFF_K);
  __bf16* Vt = (__bf16*)(ws + OFF_VT);
  __bf16* Ob = (__bf16*)(ws + OFF_O);
  __bf16* H2 = (__bf16*)(ws + OFF_H2);
  __bf16* Gb = (__bf16*)(ws + OFF_G);

  // weight conversion (N-major bf16)
  k0_convert<<<1728, 256, 0, stream>>>(qkv_w, qkvWt, CDIM, 3 * CDIM);
  k0_convert<<<576,  256, 0, stream>>>(proj_w, projWt, CDIM, CDIM);
  k0_convert<<<2304, 256, 0, stream>>>(fc1_w, fc1Wt, CDIM, MLPH);
  k0_convert<<<2304, 256, 0, stream>>>(fc2_w, fc2Wt, MLPH, CDIM);

  // LN1 + shift + partition
  k1_ln_shift<<<BNW, 256, 0, stream>>>(x, norm1_g, norm1_b, Xw);

  // QKV: M=262144 K=384 N=1152
  k2_qkv<<<dim3(2048, 18), 256, 0, stream>>>(Xw, qkvWt, qkv_b, Qb, Kb, Vt);

  // attention: one block per (window, head)
  k3_attn<<<BNW * NHEAD, 128, 0, stream>>>(Qb, Kb, Vt, rpb, Ob);

  // proj: M=262144 K=384 N=384 (writes X1 = shortcut + attn into d_out)
  k4_proj<<<dim3(2048, 6), 256, 0, stream>>>(Ob, projWt, proj_b, x, out);

  // LN2
  k5_ln2<<<NTOK / 8, 256, 0, stream>>>(out, norm2_g, norm2_b, H2);

  // FC1: M=200704 K=384 N=1536
  k6_fc1<<<dim3(1568, 24), 256, 0, stream>>>(H2, fc1Wt, fc1_b, Gb);

  // FC2: M=200704 K=1536 N=384, += into d_out
  k7_fc2<<<dim3(1568, 6), 256, 0, stream>>>(Gb, fc2Wt, fc2_b, out);
}
